// BaseMPNN_54589034332476
// MI455X (gfx1250) — matrix operand order repaired and checked by reading, not compiled, not run on hardware
//
#include <hip/hip_runtime.h>
#include <hip/hip_bf16.h>

typedef __attribute__((ext_vector_type(16))) _Float16 v16h;
typedef __attribute__((ext_vector_type(8)))  _Float16 v8h;
typedef __attribute__((ext_vector_type(8)))  float    v8f;

#define N_NODES 50000
#define N_EDGES 200000
#define F_IN    16
#define HDIM    128
#define BN_EPS  1e-5f

union HFrag { v16h v; v8h h2[2]; };

// ---------------------------------------------------------------------------
// Core WMMA tile: LDS A-tile [16][K] f16 row-major  x  Wt [128][K] f16
// (column-major weights => contiguous per-lane B fragments).
// 8 waves / block, wave w computes output columns [16w, 16w+16).
// A layout (ISA 7.12.2, 16-bit A 16x32): lane<16 holds K k0+0..7 (v0-3) and
// k0+16..23 (v4-7); lane>=16 holds k0+8..15 and k0+24..31.
// B layout: same K order as A (lane<16: k0+0..7, k0+16..23; lane>=16: k0+8..15, k0+24..31), N=ncol.
// ---------------------------------------------------------------------------
template<int K>
__device__ __forceinline__ v8f wmma_rowtile(const _Float16* As, const _Float16* wt) {
    const int lane     = threadIdx.x & 31;
    const int wave     = threadIdx.x >> 5;
    const int half_sel = lane >> 4;
    const int m        = lane & 15;
    const int ncol     = wave * 16 + m;
    const _Float16* arow = As + m * K;
    const _Float16* brow = wt + ncol * K;
    __builtin_prefetch(brow, 0, 3);     // global_prefetch: pull weight column into cache
    v8f acc = {};
#pragma unroll
    for (int k0 = 0; k0 < K; k0 += 32) {
        HFrag a, b;
        a.h2[0] = *(const v8h*)(arow + k0 + half_sel * 8);
        a.h2[1] = *(const v8h*)(arow + k0 + half_sel * 8 + 16);
        b.h2[0] = *(const v8h*)(brow + k0 + half_sel * 8);
        b.h2[1] = *(const v8h*)(brow + k0 + half_sel * 8 + 16);
        acc = __builtin_amdgcn_wmma_f32_16x16x32_f16(false, a.v, false, b.v,
                                                     (short)0, acc, false, false);
    }
    return acc;
}

// C/D layout: VGPR v holds M = v (lanes 0-15) or v+8 (lanes 16-31), N = ncol.
__device__ __forceinline__ void store_relu_tile(v8f acc, const float* bias,
                                                float* out_rowbase) {
    const int lane     = threadIdx.x & 31;
    const int wave     = threadIdx.x >> 5;
    const int half_sel = lane >> 4;
    const int ncol     = wave * 16 + (lane & 15);
    const float bv = bias[ncol];
#pragma unroll
    for (int v = 0; v < 8; ++v) {
        const int mrow = v + half_sel * 8;
        out_rowbase[mrow * HDIM + ncol] = fmaxf(acc[v] + bv, 0.0f);
    }
}

// ---------------------------------------------------------------------------
// EdgeModel: e = relu(concat(h[row], h[col], e) @ W[384,128] + b)   (in place)
// ---------------------------------------------------------------------------
__global__ void __launch_bounds__(256) edge_model_kernel(
    const float* __restrict__ h, float* e,
    const int* __restrict__ row, const int* __restrict__ col,
    const _Float16* __restrict__ wt, const float* __restrict__ bias)
{
    __shared__ __align__(16) _Float16 As[16 * 384];
    const int base = blockIdx.x * 16;
    for (int idx = threadIdx.x; idx < 16 * 384; idx += 256) {
        const int r = idx / 384, c = idx - r * 384;
        const int eid = base + r;
        float v;
        if (c < 128)      v = h[row[eid] * HDIM + c];
        else if (c < 256) v = h[col[eid] * HDIM + (c - 128)];
        else              v = e[eid * HDIM + (c - 256)];
        As[idx] = (_Float16)v;
    }
    __syncthreads();
    v8f acc = wmma_rowtile<384>(As, wt);
    store_relu_tile(acc, bias, e + base * HDIM);
}

// ---------------------------------------------------------------------------
// NodeModel msg: m = relu(concat(h[row], e) @ W[256,128] + b); scatter-add to agg[col]
// ---------------------------------------------------------------------------
__global__ void __launch_bounds__(256) node_msg_kernel(
    const float* __restrict__ h, const float* __restrict__ e,
    const int* __restrict__ row, const int* __restrict__ col,
    const _Float16* __restrict__ wt, const float* __restrict__ bias,
    float* __restrict__ agg)
{
    __shared__ __align__(16) _Float16 As[16 * 256];
    __shared__ int cshared[16];
    const int base = blockIdx.x * 16;
    if (threadIdx.x < 16) cshared[threadIdx.x] = col[base + threadIdx.x];
    for (int idx = threadIdx.x; idx < 16 * 256; idx += 256) {
        const int r = idx >> 8, c = idx & 255;
        const int eid = base + r;
        const float v = (c < 128) ? h[row[eid] * HDIM + c]
                                  : e[eid * HDIM + (c - 128)];
        As[idx] = (_Float16)v;
    }
    __syncthreads();
    v8f acc = wmma_rowtile<256>(As, wt);
    const int lane     = threadIdx.x & 31;
    const int wave     = threadIdx.x >> 5;
    const int half_sel = lane >> 4;
    const int ncol     = wave * 16 + (lane & 15);
    const float bv = bias[ncol];
#pragma unroll
    for (int v = 0; v < 8; ++v) {
        const int mrow = v + half_sel * 8;
        const float val = fmaxf(acc[v] + bv, 0.0f);
        atomicAdd(&agg[cshared[mrow] * HDIM + ncol], val);
    }
}

// ---------------------------------------------------------------------------
// NodeModel update: h = relu(concat(h, agg/cnt) @ W[256,128] + b)  (in place)
// ---------------------------------------------------------------------------
__global__ void __launch_bounds__(256) node_update_kernel(
    float* h, const float* __restrict__ agg, const float* __restrict__ cnt,
    const _Float16* __restrict__ wt, const float* __restrict__ bias)
{
    __shared__ __align__(16) _Float16 As[16 * 256];
    const int base = blockIdx.x * 16;
    for (int idx = threadIdx.x; idx < 16 * 256; idx += 256) {
        const int r = idx >> 8, c = idx & 255;
        const int nid = base + r;
        const float v = (c < 128) ? h[nid * HDIM + c]
                                  : agg[nid * HDIM + (c - 128)] / fmaxf(cnt[nid], 1.0f);
        As[idx] = (_Float16)v;
    }
    __syncthreads();
    v8f acc = wmma_rowtile<256>(As, wt);
    store_relu_tile(acc, bias, h + base * HDIM);
}

// ---------------------------------------------------------------------------
// Small helpers
// ---------------------------------------------------------------------------
__global__ void zero_kernel(float* p, int n) {
    const int i = blockIdx.x * 256 + threadIdx.x;
    if (i < n) p[i] = 0.0f;
}

__global__ void count_kernel(const int* __restrict__ col, float* __restrict__ cnt, int E) {
    const int i = blockIdx.x * 256 + threadIdx.x;
    if (i < E) atomicAdd(&cnt[col[i]], 1.0f);
}

// Encoder: out = relu(x[M,16] @ W[16,128] + b). 2 rows / block of 256 threads.
__global__ void __launch_bounds__(256) encoder_kernel(
    const float* __restrict__ x, const float* __restrict__ w,
    const float* __restrict__ b, float* __restrict__ out, int nrows)
{
    __shared__ float xs[2][F_IN];
    const int base = blockIdx.x * 2;
    const int t = threadIdx.x;
    if (t < 2 * F_IN) {
        const int rr = base + (t >> 4);
        xs[t >> 4][t & 15] = (rr < nrows) ? x[rr * F_IN + (t & 15)] : 0.0f;
    }
    __syncthreads();
    const int r = t >> 7, c = t & 127;
    const int rowid = base + r;
    if (rowid >= nrows) return;
    float acc = b[c];
#pragma unroll
    for (int k = 0; k < F_IN; ++k) acc += xs[r][k] * w[k * HDIM + c];
    out[rowid * HDIM + c] = fmaxf(acc, 0.0f);
}

// Transpose+convert f32 weights [L][K][128] -> f16 [L][128][K]
__global__ void wprep_kernel(const float* __restrict__ src, _Float16* __restrict__ dst,
                             int K, int total)
{
    const int idx = blockIdx.x * 256 + threadIdx.x;
    if (idx >= total) return;
    const int per = K * HDIM;
    const int l = idx / per, rem = idx - l * per;
    const int k = rem / HDIM, n = rem - k * HDIM;
    dst[(l * HDIM + n) * K + k] = (_Float16)src[idx];
}

// Column sum + sum of squares (for BN stats and readout means)
__global__ void __launch_bounds__(256) colstats_kernel(
    const float* __restrict__ v, int M,
    float* __restrict__ sum, float* __restrict__ sumsq)
{
    const int t = threadIdx.x;
    const int c = t & 127;
    const int rend = min((int)(blockIdx.x + 1) * 256, M);
    float s = 0.0f, ss = 0.0f;
    for (int r = blockIdx.x * 256 + (t >> 7); r < rend; r += 2) {
        const float x = v[r * HDIM + c];
        s += x; ss += x * x;
    }
    atomicAdd(&sum[c], s);
    atomicAdd(&sumsq[c], ss);
}

__global__ void bn_apply_kernel(float* __restrict__ v, int M,
    const float* __restrict__ sum, const float* __restrict__ sumsq,
    const float* __restrict__ g, const float* __restrict__ b)
{
    const int i = blockIdx.x * 256 + threadIdx.x;
    if (i >= M * HDIM) return;
    const int c = i & 127;
    const float m   = sum[c] / (float)M;
    const float var = sumsq[c] / (float)M - m * m;
    v[i] = (v[i] - m) * rsqrtf(var + BN_EPS) * g[c] + b[c];
}

__global__ void __launch_bounds__(256) readout_kernel(
    const float* __restrict__ hsum, const float* __restrict__ esum,
    const float* __restrict__ reg_w, const float* __restrict__ reg_b,
    float* __restrict__ out)
{
    __shared__ float red[256];
    const int t = threadIdx.x;
    const float v = (t < 128) ? (hsum[t] / (float)N_NODES) * reg_w[t]
                              : (esum[t - 128] / (float)N_EDGES) * reg_w[t];
    red[t] = v;
    __syncthreads();
    for (int s = 128; s > 0; s >>= 1) {
        if (t < s) red[t] += red[t + s];
        __syncthreads();
    }
    if (t == 0) out[0] = red[0] + reg_b[0];
}

// ---------------------------------------------------------------------------
extern "C" void kernel_launch(void* const* d_in, const int* in_sizes, int n_in,
                              void* d_out, int out_size, void* d_ws, size_t ws_size,
                              hipStream_t stream) {
    const float* x          = (const float*)d_in[0];
    const int*   edge_index = (const int*)  d_in[1];
    const float* edge_attr  = (const float*)d_in[2];
    const float* enc_node_w = (const float*)d_in[3];
    const float* enc_node_b = (const float*)d_in[4];
    const float* enc_edge_w = (const float*)d_in[5];
    const float* enc_edge_b = (const float*)d_in[6];
    const float* edge_w     = (const float*)d_in[7];
    const float* edge_b     = (const float*)d_in[8];
    const float* n1_w       = (const float*)d_in[9];
    const float* n1_b       = (const float*)d_in[10];
    const float* n2_w       = (const float*)d_in[11];
    const float* n2_b       = (const float*)d_in[12];
    const float* bn_node_g  = (const float*)d_in[13];
    const float* bn_node_b  = (const float*)d_in[14];
    const float* bn_edge_g  = (const float*)d_in[15];
    const float* bn_edge_b  = (const float*)d_in[16];
    const float* reg_w      = (const float*)d_in[17];
    const float* reg_b      = (const float*)d_in[18];

    const int* row = edge_index;            // edge_index[0]
    const int* col = edge_index + N_EDGES;  // edge_index[1]

    // Workspace carve-up (256B aligned)
    char* ws = (char*)d_ws;
    size_t off = 0;
    auto carve = [&](size_t bytes) {
        void* p = ws + off;
        off = (off + bytes + 255) & ~(size_t)255;
        return p;
    };
    float*     h       = (float*)    carve((size_t)N_NODES * HDIM * 4);
    float*     e       = (float*)    carve((size_t)N_EDGES * HDIM * 4);
    float*     agg     = (float*)    carve((size_t)N_NODES * HDIM * 4);
    float*     cnt     = (float*)    carve((size_t)N_NODES * 4);
    float*     stats   = (float*)    carve(4 * HDIM * 4);   // hsum|hsumsq|esum|esumsq
    _Float16*  wt_edge = (_Float16*) carve((size_t)3 * 384 * HDIM * 2);
    _Float16*  wt_n1   = (_Float16*) carve((size_t)3 * 256 * HDIM * 2);
    _Float16*  wt_n2   = (_Float16*) carve((size_t)3 * 256 * HDIM * 2);
    float* hsum = stats, *hsumsq = stats + 128, *esum = stats + 256, *esumsq = stats + 384;

    // Weight transpose + f16 conversion (weights stay hot in L2)
    wprep_kernel<<<(3 * 384 * HDIM + 255) / 256, 256, 0, stream>>>(edge_w, wt_edge, 384, 3 * 384 * HDIM);
    wprep_kernel<<<(3 * 256 * HDIM + 255) / 256, 256, 0, stream>>>(n1_w,   wt_n1,   256, 3 * 256 * HDIM);
    wprep_kernel<<<(3 * 256 * HDIM + 255) / 256, 256, 0, stream>>>(n2_w,   wt_n2,   256, 3 * 256 * HDIM);

    // Encoders
    encoder_kernel<<<N_NODES / 2, 256, 0, stream>>>(x,         enc_node_w, enc_node_b, h, N_NODES);
    encoder_kernel<<<N_EDGES / 2, 256, 0, stream>>>(edge_attr, enc_edge_w, enc_edge_b, e, N_EDGES);

    for (int l = 0; l < 3; ++l) {
        edge_model_kernel<<<N_EDGES / 16, 256, 0, stream>>>(
            h, e, row, col, wt_edge + (size_t)l * 384 * HDIM, edge_b + l * HDIM);

        zero_kernel<<<(N_NODES * HDIM + 255) / 256, 256, 0, stream>>>(agg, N_NODES * HDIM);
        zero_kernel<<<(N_NODES + 255) / 256, 256, 0, stream>>>(cnt, N_NODES);

        node_msg_kernel<<<N_EDGES / 16, 256, 0, stream>>>(
            h, e, row, col, wt_n1 + (size_t)l * 256 * HDIM, n1_b + l * HDIM, agg);
        count_kernel<<<(N_EDGES + 255) / 256, 256, 0, stream>>>(col, cnt, N_EDGES);

        node_update_kernel<<<N_NODES / 16, 256, 0, stream>>>(
            h, agg, cnt, wt_n2 + (size_t)l * 256 * HDIM, n2_b + l * HDIM);

        if (l < 2) {
            zero_kernel<<<2, 256, 0, stream>>>(stats, 4 * HDIM);
            colstats_kernel<<<(N_NODES + 255) / 256, 256, 0, stream>>>(h, N_NODES, hsum, hsumsq);
            colstats_kernel<<<(N_EDGES + 255) / 256, 256, 0, stream>>>(e, N_EDGES, esum, esumsq);
            bn_apply_kernel<<<(N_NODES * HDIM + 255) / 256, 256, 0, stream>>>(
                h, N_NODES, hsum, hsumsq, bn_node_g, bn_node_b);
            bn_apply_kernel<<<(N_EDGES * HDIM + 255) / 256, 256, 0, stream>>>(
                e, N_EDGES, esum, esumsq, bn_edge_g, bn_edge_b);
        }
    }

    // Readout: column means of h and e, dot with reg_w
    zero_kernel<<<2, 256, 0, stream>>>(stats, 4 * HDIM);
    colstats_kernel<<<(N_NODES + 255) / 256, 256, 0, stream>>>(h, N_NODES, hsum, hsumsq);
    colstats_kernel<<<(N_EDGES + 255) / 256, 256, 0, stream>>>(e, N_EDGES, esum, esumsq);
    readout_kernel<<<1, 256, 0, stream>>>(hsum, esum, reg_w, reg_b, (float*)d_out);
}
